// LlamaAttention_30614526886298
// MI455X (gfx1250) — matrix operand order repaired and checked by reading, not compiled, not run on hardware
//
#include <hip/hip_runtime.h>
#include <hip/hip_bf16.h>

// ---------------------------------------------------------------------------
// Types for gfx1250 WMMA (wave32): v16bf A/B operands, v8f f32 accumulator
// ---------------------------------------------------------------------------
typedef __attribute__((ext_vector_type(16))) __bf16 v16bf;
typedef __attribute__((ext_vector_type(8)))  __bf16 v8bf;
typedef __attribute__((ext_vector_type(8)))  float  v8f;
typedef __attribute__((ext_vector_type(4)))  unsigned short us4;

#define NH_   32
#define NKV_  8
#define HD_   128
#define QLEN_ 1024
#define PAST_ 3072
#define KVLEN_ 4096
#define HID_  4096

__device__ __forceinline__ unsigned short f2bf(float f) {
  unsigned int u = __float_as_uint(f);
  u += 0x7FFFu + ((u >> 16) & 1u);   // round-to-nearest-even
  return (unsigned short)(u >> 16);
}

__device__ __forceinline__ v16bf cat16(v8bf lo, v8bf hi) {
  return __builtin_shufflevector(lo, hi, 0,1,2,3,4,5,6,7,8,9,10,11,12,13,14,15);
}

// A operand, 16x32 bf16: lane = (g,ln): row M = ln; element e maps to
// K = e + 16*g  -> two contiguous 16B chunks.
__device__ __forceinline__ v16bf load_a16x32(const unsigned short* row, int k0, int g) {
  v8bf lo = *(const v8bf*)(row + k0 + 16 * g);
  v8bf hi = *(const v8bf*)(row + k0 + 16 * g + 8);
  return cat16(lo, hi);
}

// B operand, 32x16 bf16: lane = (g,ln): col N = ln; element e -> K = e + 16*g
// -> one contiguous 32B chunk per lane.
__device__ __forceinline__ v16bf load_b32x16(const unsigned short* base, int off) {
  v8bf lo = *(const v8bf*)(base + off);
  v8bf hi = *(const v8bf*)(base + off + 8);
  return cat16(lo, hi);
}

// ---------------------------------------------------------------------------
// fp32 -> bf16 bulk convert (4 elems / thread)
// ---------------------------------------------------------------------------
__global__ void __launch_bounds__(256)
cvt_f32_bf16(const float* __restrict__ src, unsigned short* __restrict__ dst, int n4) {
  int i = blockIdx.x * blockDim.x + threadIdx.x;
  if (i >= n4) return;
  float4 f = ((const float4*)src)[i];
  us4 o;
  o[0] = f2bf(f.x); o[1] = f2bf(f.y); o[2] = f2bf(f.z); o[3] = f2bf(f.w);
  ((us4*)dst)[i] = o;
}

// ---------------------------------------------------------------------------
// Generic GEMM: C_f32[M,N] = A_bf16[M,K] x B_bf16[N,K]^T   (nn.Linear layout)
// Block = 4 waves; each wave computes a 32(M) x 64(N) tile (8 accumulators),
// so every B fetch feeds two WMMAs (~21 flop/byte).
// ---------------------------------------------------------------------------
__global__ void __launch_bounds__(128)
gemm_bf16(const unsigned short* __restrict__ A, const unsigned short* __restrict__ Bw,
          float* __restrict__ C, int M, int N, int K) {
  const int lane = threadIdx.x & 31;
  const int wave = threadIdx.x >> 5;
  const int g = lane >> 4, ln = lane & 15;
  const int m0 = blockIdx.y * 32;
  const int n0 = blockIdx.x * 256 + wave * 64;

  const unsigned short* arow0 = A + (size_t)(m0 + ln) * K;
  const unsigned short* arow1 = A + (size_t)(m0 + 16 + ln) * K;
  const unsigned short* brow  = Bw + (size_t)(n0 + ln) * K;

  v8f acc[2][4];
#pragma unroll
  for (int i = 0; i < 2; ++i)
#pragma unroll
    for (int t = 0; t < 4; ++t) acc[i][t] = v8f{};

  for (int k0 = 0; k0 < K; k0 += 32) {
    if (k0 + 512 < K) {   // WGP-scope prefetch: A shared by 4 waves, B by M-neighbors
      __builtin_prefetch(arow0 + k0 + 512, 0, 3);
      __builtin_prefetch(arow1 + k0 + 512, 0, 3);
      __builtin_prefetch(brow + k0 + 512, 0, 3);
    }
    v16bf a0 = load_a16x32(arow0, k0, g);
    v16bf a1 = load_a16x32(arow1, k0, g);
#pragma unroll
    for (int t = 0; t < 4; ++t) {
      v16bf b = load_b32x16(brow + (size_t)t * 16 * K, k0 + 16 * g);
      acc[0][t] = __builtin_amdgcn_wmma_f32_16x16x32_bf16(false, a0, false, b, (short)0, acc[0][t], false, false);
      acc[1][t] = __builtin_amdgcn_wmma_f32_16x16x32_bf16(false, a1, false, b, (short)0, acc[1][t], false, false);
    }
  }
#pragma unroll
  for (int i = 0; i < 2; ++i) {
#pragma unroll
    for (int r = 0; r < 8; ++r) {
      size_t rowo = (size_t)(m0 + i * 16 + r + 8 * g) * N + n0;
      C[rowo + 0 * 16 + ln] = acc[i][0][r];
      C[rowo + 1 * 16 + ln] = acc[i][1][r];
      C[rowo + 2 * 16 + ln] = acc[i][2][r];
      C[rowo + 3 * 16 + ln] = acc[i][3][r];
    }
  }
}

// ---------------------------------------------------------------------------
// RoPE on Q (fold 1/sqrt(HD)); write bf16 head-major [NH][Q][HD]
// ---------------------------------------------------------------------------
__global__ void __launch_bounds__(256)
rope_q_kernel(const float* __restrict__ qf, const int* __restrict__ pos,
              unsigned short* __restrict__ qr) {
  int idx = blockIdx.x * blockDim.x + threadIdx.x;      // [s][h][d<64]
  if (idx >= QLEN_ * NH_ * 64) return;
  int d = idx & 63;
  int h = (idx >> 6) & 31;
  int s = idx >> 11;
  float inv = __powf(10000.0f, -(float)(2 * d) * (1.0f / 128.0f));
  float ang = (float)pos[s] * inv;
  float sn, c; __sincosf(ang, &sn, &c);
  const float* src = qf + (size_t)s * HID_ + h * HD_;
  float x1 = src[d], x2 = src[d + 64];
  const float scale = 0.08838834764831845f;             // 1/sqrt(128)
  unsigned short* dst = qr + ((size_t)h * QLEN_ + s) * HD_;
  dst[d]      = f2bf((x1 * c - x2 * sn) * scale);
  dst[d + 64] = f2bf((x2 * c + x1 * sn) * scale);
}

// RoPE on new K; append bf16 into k_full[NKV][KVLEN][HD] at rows PAST..
__global__ void __launch_bounds__(256)
rope_k_kernel(const float* __restrict__ kf, const int* __restrict__ pos,
              unsigned short* __restrict__ kfull) {
  int idx = blockIdx.x * blockDim.x + threadIdx.x;      // [s][kvh][d<64]
  if (idx >= QLEN_ * NKV_ * 64) return;
  int d = idx & 63;
  int kvh = (idx >> 6) & 7;
  int s = idx >> 9;
  float inv = __powf(10000.0f, -(float)(2 * d) * (1.0f / 128.0f));
  float ang = (float)pos[s] * inv;
  float sn, c; __sincosf(ang, &sn, &c);
  const float* src = kf + (size_t)s * (NKV_ * HD_) + kvh * HD_;
  float x1 = src[d], x2 = src[d + 64];
  unsigned short* dst = kfull + ((size_t)kvh * KVLEN_ + PAST_ + s) * HD_;
  dst[d]      = f2bf(x1 * c - x2 * sn);
  dst[d + 64] = f2bf(x2 * c + x1 * sn);
}

// past_k fp32 [NKV][PAST][HD] -> bf16 k_full rows 0..PAST-1 (same layout)
__global__ void __launch_bounds__(256)
append_past_k(const float* __restrict__ pk, unsigned short* __restrict__ kfull) {
  int idx = blockIdx.x * blockDim.x + threadIdx.x;
  if (idx >= NKV_ * PAST_ * HD_) return;
  int kvh = idx / (PAST_ * HD_);
  int rem = idx - kvh * (PAST_ * HD_);
  int t = rem >> 7, d = rem & 127;
  kfull[((size_t)kvh * KVLEN_ + t) * HD_ + d] = f2bf(pk[idx]);
}

// past_v fp32 [NKV][PAST][HD] -> bf16 transposed v_t[NKV][HD][KVLEN] cols 0..PAST-1
__global__ void __launch_bounds__(256)
append_past_v(const float* __restrict__ pv, unsigned short* __restrict__ vt) {
  int idx = blockIdx.x * blockDim.x + threadIdx.x;      // [kvh][d][t]
  if (idx >= NKV_ * HD_ * PAST_) return;
  int kvh = idx / (HD_ * PAST_);
  int rem = idx - kvh * (HD_ * PAST_);
  int d = rem / PAST_;
  int t = rem - d * PAST_;
  vt[((size_t)kvh * HD_ + d) * KVLEN_ + t] =
      f2bf(pv[((size_t)kvh * PAST_ + t) * HD_ + d]);
}

// new V fp32 [s][NKV*HD] -> bf16 v_t[NKV][HD][KVLEN] cols PAST..
__global__ void __launch_bounds__(256)
append_new_v(const float* __restrict__ vf, unsigned short* __restrict__ vt) {
  int idx = blockIdx.x * blockDim.x + threadIdx.x;      // [kvh][d][s]
  if (idx >= NKV_ * HD_ * QLEN_) return;
  int kvh = idx >> 17;
  int rem = idx & 131071;
  int d = rem >> 10;
  int s = rem & 1023;
  vt[((size_t)kvh * HD_ + d) * KVLEN_ + PAST_ + s] =
      f2bf(vf[(size_t)s * (NKV_ * HD_) + kvh * HD_ + d]);
}

// ---------------------------------------------------------------------------
// Flash attention: one wave per (head, 16-query tile). Online softmax fp32.
// P tile re-striped C-layout -> A-layout through LDS.
// ---------------------------------------------------------------------------
__global__ void __launch_bounds__(32)
flash_attn(const unsigned short* __restrict__ Qr,   // [NH][Q][HD] bf16 (pre-scaled)
           const unsigned short* __restrict__ Kf,   // [NKV][KVLEN][HD] bf16
           const unsigned short* __restrict__ Vt,   // [NKV][HD][KVLEN] bf16
           const float* __restrict__ mask,          // [Q][KVLEN] f32 additive
           unsigned short* __restrict__ Out) {      // [Q][NH*HD] bf16
  const int h = blockIdx.y;
  const int q0 = blockIdx.x * 16;
  const int kvh = h >> 2;                            // GROUPS = 4
  const int lane = threadIdx.x & 31;
  const int g = lane >> 4, ln = lane & 15;

  __shared__ __align__(16) unsigned short pShm[16 * 32];

  const unsigned short* qrow = Qr + ((size_t)h * QLEN_ + q0 + ln) * HD_;
  v16bf qa[4];
#pragma unroll
  for (int ks = 0; ks < 4; ++ks) qa[ks] = load_a16x32(qrow, ks * 32, g);

  v8f ot[8];
#pragma unroll
  for (int t = 0; t < 8; ++t) ot[t] = v8f{};
  float mrow[8], lrow[8];
#pragma unroll
  for (int r = 0; r < 8; ++r) { mrow[r] = -1e30f; lrow[r] = 0.0f; }

  const unsigned short* kbase = Kf + (size_t)kvh * KVLEN_ * HD_;
  const unsigned short* vbase = Vt + (size_t)kvh * HD_ * KVLEN_;

  for (int kv0 = 0; kv0 < KVLEN_; kv0 += 32) {
    const unsigned short* k0p = kbase + (size_t)(kv0 + ln) * HD_;
    const unsigned short* k1p = kbase + (size_t)(kv0 + 16 + ln) * HD_;
    if (kv0 + 32 < KVLEN_)
      __builtin_prefetch(kbase + (size_t)(kv0 + 32 + ln) * HD_, 0, 0);

    // S tiles: q[16xHD] x K^T[HDx32] -> two 16x16 f32 tiles
    v8f s0 = {}, s1 = {};
#pragma unroll
    for (int ks = 0; ks < 4; ++ks) {
      v16bf b0 = load_b32x16(k0p, ks * 32 + 16 * g);
      v16bf b1 = load_b32x16(k1p, ks * 32 + 16 * g);
      s0 = __builtin_amdgcn_wmma_f32_16x16x32_bf16(false, qa[ks], false, b0, (short)0, s0, false, false);
      s1 = __builtin_amdgcn_wmma_f32_16x16x32_bf16(false, qa[ks], false, b1, (short)0, s1, false, false);
    }

    // additive mask + online softmax (fp32), stash P into LDS (row-major 16x32)
#pragma unroll
    for (int r = 0; r < 8; ++r) {
      const float* mr = mask + (size_t)(q0 + r + 8 * g) * KVLEN_ + kv0;
      float e0 = s0[r] + mr[ln];
      float e1 = s1[r] + mr[16 + ln];
      float mx = fmaxf(e0, e1);
      mx = fmaxf(mx, __shfl_xor(mx, 1));
      mx = fmaxf(mx, __shfl_xor(mx, 2));
      mx = fmaxf(mx, __shfl_xor(mx, 4));
      mx = fmaxf(mx, __shfl_xor(mx, 8));
      float mnew = fmaxf(mrow[r], mx);
      float scale = __expf(mrow[r] - mnew);
      mrow[r] = mnew;
      float p0 = __expf(e0 - mnew);
      float p1 = __expf(e1 - mnew);
      float sum = p0 + p1;
      sum += __shfl_xor(sum, 1);
      sum += __shfl_xor(sum, 2);
      sum += __shfl_xor(sum, 4);
      sum += __shfl_xor(sum, 8);
      lrow[r] = lrow[r] * scale + sum;
#pragma unroll
      for (int t = 0; t < 8; ++t) ot[t][r] *= scale;
      pShm[(r + 8 * g) * 32 + ln] = f2bf(p0);
      pShm[(r + 8 * g) * 32 + 16 + ln] = f2bf(p1);
    }
    __syncthreads();
    v16bf pa;
    {
      const unsigned short* prow = pShm + ln * 32;
      v8bf lo = *(const v8bf*)(prow + 16 * g);
      v8bf hi = *(const v8bf*)(prow + 16 * g + 8);
      pa = cat16(lo, hi);
    }
    __syncthreads();

    // O += P[16x32] x V[32x128]
#pragma unroll
    for (int t = 0; t < 8; ++t) {
      const unsigned short* vcol = vbase + (size_t)(t * 16 + ln) * KVLEN_ + kv0 + 16 * g;
      v16bf vb = load_b32x16(vcol, 0);
      ot[t] = __builtin_amdgcn_wmma_f32_16x16x32_bf16(false, pa, false, vb, (short)0, ot[t], false, false);
    }
  }

  // epilogue: O /= l, write bf16 to [s][h*HD + d]
#pragma unroll
  for (int r = 0; r < 8; ++r) {
    float invl = 1.0f / lrow[r];
    unsigned short* orow = Out + (size_t)(q0 + r + 8 * g) * HID_ + h * HD_;
#pragma unroll
    for (int t = 0; t < 8; ++t) orow[t * 16 + ln] = f2bf(ot[t][r] * invl);
  }
}

// ---------------------------------------------------------------------------
// Orchestration
// ---------------------------------------------------------------------------
extern "C" void kernel_launch(void* const* d_in, const int* in_sizes, int n_in,
                              void* d_out, int out_size, void* d_ws, size_t ws_size,
                              hipStream_t stream) {
  (void)in_sizes; (void)n_in; (void)out_size; (void)ws_size;
  const float* hs   = (const float*)d_in[0];
  const float* mask = (const float*)d_in[1];
  const int*   pos  = (const int*)d_in[2];
  const float* pk   = (const float*)d_in[3];
  const float* pv   = (const float*)d_in[4];
  const float* Wq   = (const float*)d_in[5];
  const float* Wk   = (const float*)d_in[6];
  const float* Wv   = (const float*)d_in[7];
  const float* Wo   = (const float*)d_in[8];
  float* out = (float*)d_out;

  char* ws = (char*)d_ws;
  size_t o = 0;
  unsigned short* hsB = (unsigned short*)(ws + o); o += (size_t)QLEN_ * HID_ * 2;
  unsigned short* WqB = (unsigned short*)(ws + o); o += (size_t)HID_ * HID_ * 2;
  unsigned short* WkB = (unsigned short*)(ws + o); o += (size_t)(NKV_ * HD_) * HID_ * 2;
  unsigned short* WvB = (unsigned short*)(ws + o); o += (size_t)(NKV_ * HD_) * HID_ * 2;
  unsigned short* WoB = (unsigned short*)(ws + o); o += (size_t)HID_ * HID_ * 2;
  float* qF = (float*)(ws + o); o += (size_t)QLEN_ * HID_ * 4;
  float* kF = (float*)(ws + o); o += (size_t)QLEN_ * (NKV_ * HD_) * 4;
  float* vF = (float*)(ws + o); o += (size_t)QLEN_ * (NKV_ * HD_) * 4;
  unsigned short* qR    = (unsigned short*)(ws + o); o += (size_t)NH_ * QLEN_ * HD_ * 2;
  unsigned short* kFull = (unsigned short*)(ws + o); o += (size_t)NKV_ * KVLEN_ * HD_ * 2;
  unsigned short* vT    = (unsigned short*)(ws + o); o += (size_t)NKV_ * HD_ * KVLEN_ * 2;
  unsigned short* attnB = (unsigned short*)(ws + o); o += (size_t)QLEN_ * HID_ * 2;

  auto cvtN = [&](const float* s, unsigned short* d, int n) {
    int n4 = n / 4;
    cvt_f32_bf16<<<(n4 + 255) / 256, 256, 0, stream>>>(s, d, n4);
  };

  // 1) fp32 -> bf16 conversions
  cvtN(hs, hsB, QLEN_ * HID_);
  cvtN(Wq, WqB, HID_ * HID_);
  cvtN(Wk, WkB, (NKV_ * HD_) * HID_);
  cvtN(Wv, WvB, (NKV_ * HD_) * HID_);
  cvtN(Wo, WoB, HID_ * HID_);

  // 2) QKV projections (fp32 accumulate)
  gemm_bf16<<<dim3(HID_ / 256, QLEN_ / 32), 128, 0, stream>>>(hsB, WqB, qF, QLEN_, HID_, HID_);
  gemm_bf16<<<dim3((NKV_ * HD_) / 256, QLEN_ / 32), 128, 0, stream>>>(hsB, WkB, kF, QLEN_, NKV_ * HD_, HID_);
  gemm_bf16<<<dim3((NKV_ * HD_) / 256, QLEN_ / 32), 128, 0, stream>>>(hsB, WvB, vF, QLEN_, NKV_ * HD_, HID_);

  // 3) RoPE + KV-cache assembly
  rope_q_kernel<<<(QLEN_ * NH_ * 64) / 256, 256, 0, stream>>>(qF, pos, qR);
  rope_k_kernel<<<(QLEN_ * NKV_ * 64) / 256, 256, 0, stream>>>(kF, pos, kFull);
  append_past_k<<<(NKV_ * PAST_ * HD_) / 256, 256, 0, stream>>>(pk, kFull);
  append_past_v<<<(NKV_ * HD_ * PAST_) / 256, 256, 0, stream>>>(pv, vT);
  append_new_v<<<(NKV_ * HD_ * QLEN_) / 256, 256, 0, stream>>>(vF, vT);

  // 4) SDPA (flash, online softmax)
  flash_attn<<<dim3(QLEN_ / 16, NH_), 32, 0, stream>>>(qR, kFull, vT, mask, attnB);

  // 5) Output projection -> fp32 d_out
  gemm_bf16<<<dim3(HID_ / 256, QLEN_ / 32), 128, 0, stream>>>(attnB, WoB, out, QLEN_, HID_, HID_);
}
